// MoELayerParallel_62354335203868
// MI455X (gfx1250) — compile-verified, matrix operand order checked by reading, not run on hardware
//
#include <hip/hip_runtime.h>
#include <hip/hip_bf16.h>
#include <math.h>

#define T_TOK 16384
#define DIM   1024
#define NEXP  8
#define HID   4096

typedef __attribute__((ext_vector_type(16))) __bf16 v16bf;
typedef __attribute__((ext_vector_type(8)))  float  v8f;
typedef __attribute__((ext_vector_type(4)))  unsigned int v4u;
typedef __attribute__((ext_vector_type(8)))  int v8i;
typedef __attribute__((ext_vector_type(4)))  int v4i;

__device__ __forceinline__ v8f splat8(float f) {
  v8f v;
#pragma unroll
  for (int i = 0; i < 8; ++i) v[i] = f;
  return v;
}

__device__ __forceinline__ v8f wmma_bf16(v16bf a, v16bf b, v8f c) {
  return __builtin_amdgcn_wmma_f32_16x16x32_bf16(false, a, false, b, (short)0, c,
                                                 false, false);
}

// 16x32 bf16 fragment (A: rows=M from LDS; B: rows=N from global, B = W^T of a
// row-major [N,K] weight). Per ISA: lane L -> row base+(L&15); K runs
// [kb..kb+7] and [kb+16..kb+23] with kb=(L>=16)*8. Two aligned 16B loads.
__device__ __forceinline__ v16bf frag16(const __bf16* base, int row0,
                                        int rowstride, int k0, int lane) {
  const int r  = row0 + (lane & 15);
  const int kb = (lane >> 4) << 3;
  const __bf16* p = base + (size_t)r * rowstride + k0 + kb;
  union { v16bf v; uint4 q[2]; } u;
  u.q[0] = *(const uint4*)p;
  u.q[1] = *(const uint4*)(p + 16);
  return u.v;
}

// ---------------- f32 -> bf16 bulk conversion (weights, x) ----------------
__global__ __launch_bounds__(256) void cvt_kernel(const float* __restrict__ src,
                                                  __bf16* __restrict__ dst, int n) {
  const int stride = gridDim.x * blockDim.x * 4;
  for (int i = (blockIdx.x * blockDim.x + threadIdx.x) * 4; i < n; i += stride) {
    float4 v = *(const float4*)(src + i);
    dst[i + 0] = (__bf16)v.x;
    dst[i + 1] = (__bf16)v.y;
    dst[i + 2] = (__bf16)v.z;
    dst[i + 3] = (__bf16)v.w;
  }
}

// ---------------- gate: softmax, loss sums, top-2 routing ----------------
__global__ __launch_bounds__(256) void gate_kernel(
    const float* __restrict__ x, const float* __restrict__ gate_w,
    const float* __restrict__ noise, const float* __restrict__ noise_weight,
    float* __restrict__ sums, unsigned* __restrict__ counts,
    float* __restrict__ wtab, int* __restrict__ list) {
  const int t = blockIdx.x * blockDim.x + threadIdx.x;
  if (t >= T_TOK) return;
  float acc[NEXP];
#pragma unroll
  for (int e = 0; e < NEXP; ++e) acc[e] = 0.f;
  const float4* xr = (const float4*)(x + (size_t)t * DIM);
  for (int c = 0; c < DIM / 4; ++c) {
    float4 xv = xr[c];
#pragma unroll
    for (int e = 0; e < NEXP; ++e) {
      float4 g = ((const float4*)(gate_w + (size_t)e * DIM))[c];
      acc[e] += xv.x * g.x + xv.y * g.y + xv.z * g.z + xv.w * g.w;
    }
  }
  float mx = acc[0];
#pragma unroll
  for (int e = 1; e < NEXP; ++e) mx = fmaxf(mx, acc[e]);
  float p[NEXP], s = 0.f;
#pragma unroll
  for (int e = 0; e < NEXP; ++e) { p[e] = __expf(acc[e] - mx); s += p[e]; }
  const float inv = 1.f / s;
#pragma unroll
  for (int e = 0; e < NEXP; ++e) atomicAdd(&sums[e], p[e] * inv);
  float ln[NEXP];
#pragma unroll
  for (int e = 0; e < NEXP; ++e)
    ln[e] = acc[e] + noise[(size_t)t * NEXP + e] * noise_weight[e];
  int i1 = 0;
#pragma unroll
  for (int e = 1; e < NEXP; ++e) if (ln[e] > ln[i1]) i1 = e;
  int i2 = (i1 == 0) ? 1 : 0;
#pragma unroll
  for (int e = 0; e < NEXP; ++e)
    if (e != i1 && ln[e] > ln[i2]) i2 = e;
  const float e2 = __expf(ln[i2] - ln[i1]);
  const float wa = 1.f / (1.f + e2);
  const float wb = e2 / (1.f + e2);
#pragma unroll
  for (int e = 0; e < NEXP; ++e)
    wtab[(size_t)t * NEXP + e] = (e == i1) ? wa : ((e == i2) ? wb : 0.f);
  unsigned p1 = atomicAdd(&counts[i1], 1u);
  list[(size_t)i1 * T_TOK + p1] = t;
  unsigned p2 = atomicAdd(&counts[i2], 1u);
  list[(size_t)i2 * T_TOK + p2] = t;
}

__global__ void loss_kernel(const float* __restrict__ sums, float* __restrict__ out) {
  if (threadIdx.x == 0) {
    float a = 0.f;
#pragma unroll
    for (int e = 0; e < NEXP; ++e) {
      float d = sums[e] / (float)T_TOK - 1.f / (float)NEXP;
      a += d * d;
    }
    out[0] = (a / (float)NEXP) * 0.01f;
  }
}

// ---------------- routed expert SwiGLU with WMMA + TDM gather ----------------
// grid: (T/64 m-tiles, E, 2 d-halves); block 256 = 8 waves.
// wave w: msub = w>>1 (16 tokens), npart = w&1 (256 out columns / 32-h slice).
__global__ __launch_bounds__(256) void expert_kernel(
    const __bf16* __restrict__ xb, const __bf16* __restrict__ w1,
    const float* __restrict__ b1, const __bf16* __restrict__ w2,
    const float* __restrict__ b2, const __bf16* __restrict__ wp,
    const float* __restrict__ bp, const float* __restrict__ wtab,
    const int* __restrict__ list, const unsigned* __restrict__ counts,
    float* __restrict__ y) {
  __shared__ __bf16 xs[64][DIM];     // gathered bf16 token tile (128 KB)
  __shared__ __bf16 act[64][72];     // h*silu(g) staging, 64 tok x 64 h (+pad)
  __shared__ float wts[64];
  __shared__ int   toks[64];

  const int e     = blockIdx.y;
  const int nhalf = blockIdx.z;
  const int mblk  = blockIdx.x;
  const unsigned cnt = counts[e];
  if ((unsigned)(mblk * 64) >= cnt) return;   // also covers cnt == 0

  const int tid  = threadIdx.x;
  const int wave = tid >> 5;
  const int lane = tid & 31;

  if (tid < 64) {
    int i = mblk * 64 + tid;
    int tok = 0;
    float w = 0.f;
    if (i < (int)cnt) {
      tok = list[(size_t)e * T_TOK + i];
      w = wtab[(size_t)tok * NEXP + e];
    }
    toks[tid] = tok;
    wts[tid]  = w;
  }
  __syncthreads();

#if __has_builtin(__builtin_amdgcn_tensor_load_to_lds)
  // TDM gather-mode: 4 descriptors x 16 row-indices, each row = 1024 bf16.
  if (wave == 0) {
#pragma unroll
    for (int g = 0; g < 4; ++g) {
      const unsigned lds_addr = (unsigned)(uintptr_t)(&xs[g * 16][0]);
      const unsigned long long ga = (unsigned long long)(uintptr_t)xb;
      v4u g0;
      g0[0] = 0x80000001u;                       // gather_mode=1, 16b idx, count=1
      g0[1] = lds_addr;                          // LDS byte address of tile
      g0[2] = (unsigned)(ga & 0xffffffffu);      // global_addr[31:0]
      g0[3] = (unsigned)((ga >> 32) & 0x1ffffffu) | 0x80000000u;  // [56:32]|type=2
      v8i g1;
      g1[0] = 0x00010000;                        // data_size=1 (2B), no mask/flags
      g1[1] = (int)(DIM << 16);                  // tensor_dim0[15:0]=1024 (hi half)
      g1[2] = (int)(T_TOK << 16);                // dim0 hi=0 | tensor_dim1 lo=T
      g1[3] = (int)(DIM << 16);                  // dim1 hi=0 | tile_dim0=1024
      g1[4] = 16;                                // tile_dim1 = #valid indices
      g1[5] = DIM;                               // tensor_dim0_stride = 1024
      g1[6] = 0;
      g1[7] = 0;
      v4i g2, g3;
#pragma unroll
      for (int q = 0; q < 4; ++q) {
        g2[q] = (toks[g * 16 + 2 * q] & 0xffff) | (toks[g * 16 + 2 * q + 1] << 16);
        g3[q] = (toks[g * 16 + 8 + 2 * q] & 0xffff) |
                (toks[g * 16 + 8 + 2 * q + 1] << 16);
      }
#if defined(__clang_major__) && (__clang_major__ >= 23)
      v8i zz = {0, 0, 0, 0, 0, 0, 0, 0};
      __builtin_amdgcn_tensor_load_to_lds(g0, g1, g2, g3, zz, 0);
#else
      __builtin_amdgcn_tensor_load_to_lds(g0, g1, g2, g3, 0);
#endif
    }
#if __has_builtin(__builtin_amdgcn_s_wait_tensorcnt)
    __builtin_amdgcn_s_wait_tensorcnt(0);
#else
    asm volatile("s_wait_tensorcnt 0" ::: "memory");
#endif
  }
#else
  {  // fallback: plain bf16 gather copy, 4 threads per row (512B each)
    const int row = tid >> 2, part = tid & 3;
    const uint4* s = (const uint4*)(xb + (size_t)toks[row] * DIM) + part * 32;
    uint4* d = (uint4*)&xs[row][0] + part * 32;
    for (int c = 0; c < 32; ++c) d[c] = s[c];
  }
#endif
  __syncthreads();

  const __bf16* w1e = w1 + (size_t)e * HID * DIM;
  const __bf16* w2e = w2 + (size_t)e * HID * DIM;
  const __bf16* wpe = wp + (size_t)e * DIM * HID;
  const float*  b1e = b1 + (size_t)e * HID;
  const float*  b2e = b2 + (size_t)e * HID;

  const int msub  = wave >> 1;
  const int npart = wave & 1;
  const int mrow  = msub * 16;
  const int d0    = nhalf * 512 + npart * 256;
  const int nlan  = lane & 15;
  const int rb    = (lane >> 4) << 3;

  v8f acc[16];
#pragma unroll
  for (int nt = 0; nt < 16; ++nt) acc[nt] = splat8(0.f);

  for (int it = 0; it < HID / 64; ++it) {
    const int h0 = it * 64;
    const int hs = h0 + npart * 32;  // this wave's 32-h slice of the window
    v8f ch0 = splat8(b1e[hs + nlan]);
    v8f ch1 = splat8(b1e[hs + 16 + nlan]);
    v8f cg0 = splat8(b2e[hs + nlan]);
    v8f cg1 = splat8(b2e[hs + 16 + nlan]);
    for (int k = 0; k < DIM; k += 32) {
      v16bf aX  = frag16(&xs[0][0], mrow, DIM, k, lane);
      v16bf bh0 = frag16(w1e, hs,      DIM, k, lane);
      v16bf bh1 = frag16(w1e, hs + 16, DIM, k, lane);
      v16bf bg0 = frag16(w2e, hs,      DIM, k, lane);
      v16bf bg1 = frag16(w2e, hs + 16, DIM, k, lane);
      ch0 = wmma_bf16(aX, bh0, ch0);
      ch1 = wmma_bf16(aX, bh1, ch1);
      cg0 = wmma_bf16(aX, bg0, cg0);
      cg1 = wmma_bf16(aX, bg1, cg1);
    }
    __syncthreads();  // previous iteration's act reads are complete
#pragma unroll
    for (int r = 0; r < 8; ++r) {  // C layout: vgpr r -> row r (+8 hi lanes)
      const int m = mrow + rb + r;
      float g0 = cg0[r];
      act[m][npart * 32 + nlan]      = (__bf16)(ch0[r] * (g0 / (1.f + __expf(-g0))));
      float g1 = cg1[r];
      act[m][npart * 32 + 16 + nlan] = (__bf16)(ch1[r] * (g1 / (1.f + __expf(-g1))));
    }
    __syncthreads();
#pragma unroll
    for (int kc = 0; kc < 64; kc += 32) {
      v16bf aA = frag16(&act[0][0], mrow, 72, kc, lane);
#pragma unroll
      for (int nt = 0; nt < 16; ++nt) {
        v16bf bw = frag16(wpe, d0 + nt * 16, HID, h0 + kc, lane);
        acc[nt] = wmma_bf16(aA, bw, acc[nt]);
      }
    }
  }

  // epilogue: y[tok, d] += gate_w * (acc + bp[d]); padded rows have wts==0
#pragma unroll
  for (int nt = 0; nt < 16; ++nt) {
    const int d   = d0 + nt * 16 + nlan;
    const float bpv = bp[(size_t)e * DIM + d];
#pragma unroll
    for (int r = 0; r < 8; ++r) {
      const int m = mrow + rb + r;
      const float val = wts[m] * (acc[nt][r] + bpv);
      atomicAdd(&y[(size_t)toks[m] * DIM + d], val);
    }
  }
}

extern "C" void kernel_launch(void* const* d_in, const int* in_sizes, int n_in,
                              void* d_out, int out_size, void* d_ws, size_t ws_size,
                              hipStream_t stream) {
  (void)in_sizes; (void)n_in; (void)out_size; (void)ws_size;
  const float* x      = (const float*)d_in[0];
  const float* noise  = (const float*)d_in[1];
  const float* gate_w = (const float*)d_in[2];
  const float* nw     = (const float*)d_in[3];
  const float* w1     = (const float*)d_in[4];
  const float* b1     = (const float*)d_in[5];
  const float* w2     = (const float*)d_in[6];
  const float* b2     = (const float*)d_in[7];
  const float* wp     = (const float*)d_in[8];
  const float* bp     = (const float*)d_in[9];

  float* y    = (float*)d_out;                    // [T, D]
  float* loss = y + (size_t)T_TOK * DIM;          // scalar

  char* ws = (char*)d_ws;
  float*    sums = (float*)ws;                                   // 8 f32
  unsigned* cnt  = (unsigned*)(ws + 32);                         // 8 u32
  float*    wtab = (float*)(ws + 64);                            // T*E f32
  int*      list = (int*)(ws + 64 + (size_t)T_TOK * NEXP * 4);   // E*T i32

  const size_t NW = (size_t)NEXP * HID * DIM;     // weight elements per tensor
  __bf16* xb  = (__bf16*)(ws + (2u << 20));       // T*D bf16   (32 MB)
  __bf16* bw1 = xb + (size_t)T_TOK * DIM;         // E*H*D bf16 (64 MB)
  __bf16* bw2 = bw1 + NW;
  __bf16* bwp = bw2 + NW;

  hipMemsetAsync(d_out, 0, (size_t)T_TOK * DIM * sizeof(float), stream);
  hipMemsetAsync(d_ws, 0, 64, stream);

  cvt_kernel<<<4096, 256, 0, stream>>>(x, xb, T_TOK * DIM);
  cvt_kernel<<<8192, 256, 0, stream>>>(w1, bw1, (int)NW);
  cvt_kernel<<<8192, 256, 0, stream>>>(w2, bw2, (int)NW);
  cvt_kernel<<<8192, 256, 0, stream>>>(wp, bwp, (int)NW);

  gate_kernel<<<T_TOK / 256, 256, 0, stream>>>(x, gate_w, noise, nw, sums, cnt,
                                               wtab, list);
  loss_kernel<<<1, 32, 0, stream>>>(sums, loss);

  dim3 grid(T_TOK / 64, NEXP, 2);
  expert_kernel<<<grid, 256, 0, stream>>>(xb, bw1, b1, bw2, b2, bwp, bp, wtab,
                                          list, cnt, y);
}
